// RNNClassifier_49031346651246
// MI455X (gfx1250) — matrix operand order repaired and checked by reading, not compiled, not run on hardware
//
#include <hip/hip_runtime.h>

// Problem sizes (fixed by the reference)
#define VV   50000
#define EE   256
#define HH   512
#define TT   50
#define CC   5
#define BB   128
#define SS   512
#define FC1N 500
#define G4   2048   // 4*H

typedef __attribute__((ext_vector_type(16))) __bf16 v16bf;
typedef __attribute__((ext_vector_type(8)))  float  v8f;

union Frag16 {
    v16bf v;
    unsigned short u[16];
    uint4 q[2];
};

__device__ __forceinline__ unsigned short f2bf(float f) {
    unsigned u = __float_as_uint(f);
    unsigned r = (u >> 16) & 1u;            // round-to-nearest-even
    return (unsigned short)((u + 0x7FFFu + r) >> 16);
}
__device__ __forceinline__ float sigf(float x) { return 1.0f / (1.0f + __expf(-x)); }

// ---------------------------------------------------------------------------
// prep kernels
// ---------------------------------------------------------------------------
__global__ void zero_counters_kernel(unsigned* counters) {
    if (threadIdx.x < SS) counters[threadIdx.x] = 0u;
}

__global__ void convert_f32_bf16_kernel(const float* __restrict__ src,
                                        unsigned short* __restrict__ dst, int n) {
    int i = blockIdx.x * blockDim.x + threadIdx.x;
    if (i < n) dst[i] = f2bf(src[i]);
}

// tbias[b][g] = bias[g] + sum_t topic[b,t] * W_t[t,g]
__global__ void tbias_kernel(const float* __restrict__ topic,
                             const float* __restrict__ W_t,
                             const float* __restrict__ bias,
                             float* __restrict__ tb) {
    int i = blockIdx.x * blockDim.x + threadIdx.x;
    if (i >= BB * G4) return;
    int b = i / G4, g = i % G4;
    float acc = bias[g];
    for (int t = 0; t < TT; ++t) acc = fmaf(topic[b * TT + t], W_t[t * G4 + g], acc);
    tb[i] = acc;
}

// ---------------------------------------------------------------------------
// persistent LSTM kernel: 32 blocks (one per 16-col H tile), 8 waves/block
// (one per 16-row batch tile). c state lives in registers; W slices in LDS.
// Split-phase grid barrier: the h-independent embed@W_x GEMM runs *between*
// signaling step s-1 and waiting on it, hiding the inter-WGP sync latency.
// ---------------------------------------------------------------------------
template <bool EMBBF>
__global__ void lstm_kernel(const int*   __restrict__ x,         // [B,S]
                            const float* __restrict__ embed_f,   // [V,E] fp32
                            const unsigned short* __restrict__ embed_bf, // [V,E] bf16
                            const float* __restrict__ c0,        // [B,H]
                            const unsigned short* __restrict__ wx_bf, // [E,4H]
                            const unsigned short* __restrict__ wh_bf, // [H,4H]
                            const float* __restrict__ tbias,     // [B,4H]
                            unsigned short* hb0,                 // [B,H] bf16 ping
                            unsigned short* hb1,                 // [B,H] bf16 pong
                            float* __restrict__ h_last,          // [B,H]
                            unsigned* counters)                  // [S]
{
    extern __shared__ char smem[];
    unsigned short* whl = (unsigned short*)smem;                 // [64][512]
    unsigned short* wxl = (unsigned short*)(smem + 64 * HH * 2); // [64][256]

    const int jt   = blockIdx.x;         // H-column tile (0..31)
    const int tid  = threadIdx.x;
    const int lane = tid & 31;
    const int m    = tid >> 5;           // wave == batch-row tile (0..7)
    const int n    = lane & 15;          // output column within tile
    const int hf   = lane >> 4;          // lane-half (WMMA layout selector)

    // ---- stage this block's 64 weight columns (4 gates x 16 cols) into LDS ----
    for (int e = tid; e < 64 * HH; e += blockDim.x) {
        int cl = e >> 9, k = e & (HH - 1);
        int col = (cl >> 4) * HH + jt * 16 + (cl & 15);
        whl[cl * HH + k] = wh_bf[k * G4 + col];
    }
    for (int e = tid; e < 64 * EE; e += blockDim.x) {
        int cl = e >> 8, k = e & (EE - 1);
        int col = (cl >> 4) * HH + jt * 16 + (cl & 15);
        wxl[cl * EE + k] = wx_bf[k * G4 + col];
    }
    __syncthreads();

    // ---- preload per-wave constants: topic bias fragments + c state ----
    v8f tb[4], c;
    #pragma unroll
    for (int g = 0; g < 4; ++g)
        #pragma unroll
        for (int r = 0; r < 8; ++r) {
            int row = m * 16 + r + 8 * hf;
            tb[g][r] = tbias[row * G4 + g * HH + jt * 16 + n];
        }
    #pragma unroll
    for (int r = 0; r < 8; ++r) {
        int row = m * 16 + r + 8 * hf;
        c[r] = c0[row * HH + jt * 16 + n];
    }

    const int arow = m * 16 + n;         // A-matrix row owned by this lane
    const int* xrow = x + arow * SS;
    unsigned short* hcur = hb0;
    unsigned short* hnxt = hb1;

    for (int s = 0; s < SS; ++s) {
        // token for this step, plus prefetch of next step's embed row
        int tok = xrow[s];
        if (s + 1 < SS) {
            int tok_n = xrow[s + 1];
            if (EMBBF) {
                const char* pn = (const char*)(embed_bf + (long)tok_n * EE);
                __builtin_prefetch(pn, 0, 1);
                __builtin_prefetch(pn + 256, 0, 1);
            } else {
                const char* pn = (const char*)(embed_f + (long)tok_n * EE);
                __builtin_prefetch(pn, 0, 1);
                __builtin_prefetch(pn + 512, 0, 1);
            }
        }

        v8f acc[4];
        #pragma unroll
        for (int g = 0; g < 4; ++g) acc[g] = tb[g];

        // ------- embed[x_s] @ W_x (K=256): independent of h(s-1) -------------
        // runs in the shadow of the other blocks finishing step s-1
        #pragma unroll 2
        for (int ks = 0; ks < EE / 32; ++ks) {
            int kb = ks * 32;
            Frag16 a;
            if (EMBBF) {
                const unsigned short* erow = embed_bf + (long)tok * EE;
                a.q[0] = *(const uint4*)(erow + kb + 8 * hf);
                a.q[1] = *(const uint4*)(erow + kb + 16 + 8 * hf);
            } else {
                const float* erow = embed_f + (long)tok * EE;
                float4 f0 = *(const float4*)(erow + kb + 8 * hf);
                float4 f1 = *(const float4*)(erow + kb + 8 * hf + 4);
                float4 f2 = *(const float4*)(erow + kb + 16 + 8 * hf);
                float4 f3 = *(const float4*)(erow + kb + 16 + 8 * hf + 4);
                a.u[0]  = f2bf(f0.x); a.u[1]  = f2bf(f0.y); a.u[2]  = f2bf(f0.z); a.u[3]  = f2bf(f0.w);
                a.u[4]  = f2bf(f1.x); a.u[5]  = f2bf(f1.y); a.u[6]  = f2bf(f1.z); a.u[7]  = f2bf(f1.w);
                a.u[8]  = f2bf(f2.x); a.u[9]  = f2bf(f2.y); a.u[10] = f2bf(f2.z); a.u[11] = f2bf(f2.w);
                a.u[12] = f2bf(f3.x); a.u[13] = f2bf(f3.y); a.u[14] = f2bf(f3.z); a.u[15] = f2bf(f3.w);
            }
            Frag16 bf[4];
            #pragma unroll
            for (int g = 0; g < 4; ++g) {
                const unsigned short* wp = wxl + (g * 16 + n) * EE + kb + 8 * hf;
                bf[g].q[0] = *(const uint4*)(wp);
                bf[g].q[1] = *(const uint4*)(wp + 16);
            }
            #pragma unroll
            for (int g = 0; g < 4; ++g)
                acc[g] = __builtin_amdgcn_wmma_f32_16x16x32_bf16(
                    false, a.v, false, bf[g].v, (short)0, acc[g], false, false);
        }

        // ------- WAIT phase: h(s-1) must be globally visible ------------------
        if (s > 0) {
            if (tid == 0) {
                while (__hip_atomic_load(counters + (s - 1), __ATOMIC_ACQUIRE,
                                         __HIP_MEMORY_SCOPE_AGENT) < (unsigned)gridDim.x) {
                    __builtin_amdgcn_s_sleep(1);
                }
            }
            __syncthreads();
            __threadfence();            // acquire for all threads' subsequent reads
        }

        // ------- h(s-1) @ W_h (K=512) ----------------------------------------
        const unsigned short* hrow = hcur + arow * HH;
        #pragma unroll 2
        for (int ks = 0; ks < HH / 32; ++ks) {
            int kb = ks * 32;
            Frag16 a;
            a.q[0] = *(const uint4*)(hrow + kb + 8 * hf);        // K: kb+8h .. +7
            a.q[1] = *(const uint4*)(hrow + kb + 16 + 8 * hf);   // K: kb+16+8h .. +7
            Frag16 bf[4];
            #pragma unroll
            for (int g = 0; g < 4; ++g) {
                const unsigned short* wp = whl + (g * 16 + n) * HH + kb + 8 * hf;
                bf[g].q[0] = *(const uint4*)(wp);
                bf[g].q[1] = *(const uint4*)(wp + 16);
            }
            #pragma unroll
            for (int g = 0; g < 4; ++g)
                acc[g] = __builtin_amdgcn_wmma_f32_16x16x32_bf16(
                    false, a.v, false, bf[g].v, (short)0, acc[g], false, false);
        }

        // ------- gates, state update -----------------------------------------
        v8f h;
        #pragma unroll
        for (int r = 0; r < 8; ++r) {
            float ig = sigf(acc[0][r]);
            float fg = sigf(acc[1][r]);
            float gg = tanhf(acc[2][r]);
            float og = sigf(acc[3][r]);
            float cn = fg * c[r] + ig * gg;
            c[r] = cn;
            h[r] = og * tanhf(cn);
        }
        #pragma unroll
        for (int r = 0; r < 8; ++r) {
            int row = m * 16 + r + 8 * hf;
            hnxt[row * HH + jt * 16 + n] = f2bf(h[r]);
        }
        if (s == SS - 1) {
            #pragma unroll
            for (int r = 0; r < 8; ++r) {
                int row = m * 16 + r + 8 * hf;
                h_last[row * HH + jt * 16 + n] = h[r];
            }
        }

        // ------- SIGNAL phase: publish h(s) ----------------------------------
        __threadfence();                // release: each thread's h stores
        __syncthreads();
        if (tid == 0) {
            __hip_atomic_fetch_add(counters + s, 1u, __ATOMIC_ACQ_REL,
                                   __HIP_MEMORY_SCOPE_AGENT);
        }

        unsigned short* t = hcur; hcur = hnxt; hnxt = t;
    }
}

// ---------------------------------------------------------------------------
// MLP head (tiny: keep exact fp32)
// ---------------------------------------------------------------------------
__global__ void fc1_kernel(const float* __restrict__ hl, const float* __restrict__ w,
                           const float* __restrict__ bias, float* __restrict__ hid) {
    int i = blockIdx.x * blockDim.x + threadIdx.x;
    if (i >= BB * FC1N) return;
    int b = i / FC1N, k = i % FC1N;
    const float* hr = hl + b * HH;
    const float* wr = w + k * HH;
    float acc = bias[k];
    for (int j = 0; j < HH; ++j) acc = fmaf(hr[j], wr[j], acc);
    hid[i] = acc;
}

__global__ void fc2_kernel(const float* __restrict__ hid, const float* __restrict__ w,
                           const float* __restrict__ bias, float* __restrict__ out) {
    int i = blockIdx.x * blockDim.x + threadIdx.x;
    if (i >= BB * CC) return;
    int b = i / CC, cc = i % CC;
    const float* hr = hid + b * FC1N;
    const float* wr = w + cc * FC1N;
    float acc = bias[cc];
    for (int j = 0; j < FC1N; ++j) acc = fmaf(hr[j], wr[j], acc);
    out[i] = acc;
}

// ---------------------------------------------------------------------------
extern "C" void kernel_launch(void* const* d_in, const int* in_sizes, int n_in,
                              void* d_out, int out_size, void* d_ws, size_t ws_size,
                              hipStream_t stream) {
    const int*   x     = (const int*)d_in[0];
    const float* topic = (const float*)d_in[1];
    const float* h0    = (const float*)d_in[2];
    const float* c0    = (const float*)d_in[3];
    const float* embed = (const float*)d_in[4];
    const float* W_x   = (const float*)d_in[5];
    const float* W_t   = (const float*)d_in[6];
    const float* W_h   = (const float*)d_in[7];
    const float* bvec  = (const float*)d_in[8];
    const float* fc1_w = (const float*)d_in[9];
    const float* fc1_b = (const float*)d_in[10];
    const float* fc2_w = (const float*)d_in[11];
    const float* fc2_b = (const float*)d_in[12];
    float* logits = (float*)d_out;

    // workspace layout (base < 5 MB; optional bf16 embed copy adds 25.6 MB)
    char* ws = (char*)d_ws;
    size_t off = 0;
    unsigned*       counters = (unsigned*)(ws + off);        off += 4096;
    unsigned short* wx_bf    = (unsigned short*)(ws + off);  off += (size_t)EE * G4 * 2;
    unsigned short* wh_bf    = (unsigned short*)(ws + off);  off += (size_t)HH * G4 * 2;
    float*          tbias    = (float*)(ws + off);           off += (size_t)BB * G4 * 4;
    unsigned short* hb0      = (unsigned short*)(ws + off);  off += (size_t)BB * HH * 2;
    unsigned short* hb1      = (unsigned short*)(ws + off);  off += (size_t)BB * HH * 2;
    float*          h_last   = (float*)(ws + off);           off += (size_t)BB * HH * 4;
    float*          hid      = (float*)(ws + off);           off += 262144;
    unsigned short* embed_bf = (unsigned short*)(ws + off);
    size_t need_embbf = off + (size_t)VV * EE * 2;
    const bool use_embbf = (ws_size >= need_embbf);

    zero_counters_kernel<<<1, SS, 0, stream>>>(counters);
    convert_f32_bf16_kernel<<<(EE * G4 + 255) / 256, 256, 0, stream>>>(W_x, wx_bf, EE * G4);
    convert_f32_bf16_kernel<<<(HH * G4 + 255) / 256, 256, 0, stream>>>(W_h, wh_bf, HH * G4);
    convert_f32_bf16_kernel<<<(BB * HH + 255) / 256, 256, 0, stream>>>(h0, hb0, BB * HH);
    if (use_embbf)
        convert_f32_bf16_kernel<<<(VV * EE + 255) / 256, 256, 0, stream>>>(embed, embed_bf, VV * EE);
    tbias_kernel<<<(BB * G4 + 255) / 256, 256, 0, stream>>>(topic, W_t, bvec, tbias);

    // 32 blocks (co-resident, one per H-column tile) x 8 waves; 96 KB dynamic LDS
    const unsigned lds_bytes = 64 * HH * 2 + 64 * EE * 2;
    if (use_embbf) {
        lstm_kernel<true><<<32, 256, lds_bytes, stream>>>(
            x, embed, embed_bf, c0, wx_bf, wh_bf, tbias, hb0, hb1, h_last, counters);
    } else {
        lstm_kernel<false><<<32, 256, lds_bytes, stream>>>(
            x, embed, embed_bf, c0, wx_bf, wh_bf, tbias, hb0, hb1, h_last, counters);
    }

    fc1_kernel<<<(BB * FC1N + 255) / 256, 256, 0, stream>>>(h_last, fc1_w, fc1_b, hid);
    fc2_kernel<<<(BB * CC + 63) / 64, 64, 0, stream>>>(hid, fc2_w, fc2_b, logits);
}
